// GNNCASimple_22179211117295
// MI455X (gfx1250) — matrix operand order repaired and checked by reading, not compiled, not run on hardware
//
#include <hip/hip_runtime.h>

// ---------------------------------------------------------------------------
// GNNCASimple for MI455X (gfx1250): f16 WMMA GEMMs + f32-atomic edge scatter.
// Round 2: hoist A fragments out of the nt loop (registerized via template KT)
// to cut per-layer LDS reads 8x and keep the WMMA pipe fed from registers.
// ---------------------------------------------------------------------------

typedef _Float16 half_t;
typedef _Float16 v16h __attribute__((ext_vector_type(16)));
typedef _Float16 v8h  __attribute__((ext_vector_type(8)));
typedef _Float16 v4h  __attribute__((ext_vector_type(4)));
typedef float    v8f  __attribute__((ext_vector_type(8)));

#define N_NODES 100000
#define N_EDGES 1600000
#define DIM 16
#define H   128
#define WAVES 4   // waves (16-node tiles) per block

union V16H { v16h v; v8h h[2]; };

__device__ __forceinline__ v8f wmma_f16(v16h a, v16h b, v8f c) {
    // (neg_a, A, neg_b, B, c_mod, C, reuse_a, reuse_b)
    return __builtin_amdgcn_wmma_f32_16x16x32_f16(false, a, false, b, (short)0, c,
                                                  false, false);
}

// ---- B fragment: pre-swizzled f16 weights, 32 contiguous bytes per lane ----
__device__ __forceinline__ v16h load_B_frag(const half_t* __restrict__ Wf,
                                            int kt, int nt, int NT, int lane) {
    const v8h* p = (const v8h*)(Wf + (((size_t)(kt * NT + nt) << 9) + ((size_t)lane << 4)));
    V16H b; b.h[0] = p[0]; b.h[1] = p[1];
    return b.v;
}

// ---- A fragment from an LDS tile stored row-major [16][ldK] f16 ----
// ISA 16-bit A 16x32 layout: lanes 0-15 hold K = e (e<8) / e+8 (e>=8);
// lanes 16-31 hold K = e+8 (e<8) / e+16 (e>=8)  => two 16B contiguous loads.
__device__ __forceinline__ v16h load_A_lds(const half_t* lds, int lane, int kbase, int ldK) {
    const int m   = lane & 15;
    const int off = kbase + ((lane < 16) ? 0 : 8);
    V16H a;
    a.h[0] = *(const v8h*)(lds + m * ldK + off);
    a.h[1] = *(const v8h*)(lds + m * ldK + off + 16);
    return a.v;
}

// ---- C tile (v8f) -> LDS f16 with bias + ReLU.  C layout: VGPR r holds
// row (r + 8*(lane>=16)), col = lane&15 within the 16x16 N-tile. ----
__device__ __forceinline__ void store_C_lds(half_t* lds, int lane, int nt, int ldK,
                                            v8f c, const float* __restrict__ bias,
                                            bool do_relu) {
    const int col   = nt * 16 + (lane & 15);
    const int mbase = (lane < 16) ? 0 : 8;
    const float bv  = bias[col];
#pragma unroll
    for (int r = 0; r < 8; ++r) {
        float v = c[r] + bv;
        if (do_relu) v = fmaxf(v, 0.0f);
        lds[(mbase + r) * ldK + col] = (half_t)v;
    }
}

// A fragments loaded ONCE into registers (KT*8 VGPRs), reused across all nt.
template <int KT, int NT>
__device__ __forceinline__ void gemm_layer(const half_t* Asrc, half_t* Adst,
                                           const half_t* __restrict__ Wf,
                                           const float* __restrict__ bias,
                                           int lane, int ldKin, int ldKout, bool relu) {
    v16h afr[KT];
#pragma unroll
    for (int kt = 0; kt < KT; ++kt) afr[kt] = load_A_lds(Asrc, lane, kt * 32, ldKin);
#pragma unroll
    for (int nt = 0; nt < NT; ++nt) {
        v8f c = {};
#pragma unroll
        for (int kt = 0; kt < KT; ++kt)
            c = wmma_f16(afr[kt], load_B_frag(Wf, kt, nt, NT, lane), c);
        store_C_lds(Adst, lane, nt, ldKout, c, bias, relu);
    }
    asm volatile("s_wait_dscnt 0" ::: "memory");   // cross-lane LDS RAW fence
}

// ---------------------------------------------------------------------------
// Kernel 0: swizzle an f32 weight [K x Ncols] into f16 WMMA B fragments.
// frag(kt,nt) = 512 halves; element (lane*16+e) = W[kt*32 + 8*(lane>=16) + e + 8*(e>=8)][nt*16 + lane&15]
// Rows >= Kreal are zero-padded (used for W1: K=16 padded to 32).
// ---------------------------------------------------------------------------
__global__ void gnnca_swizzle_w(const float* __restrict__ src, half_t* __restrict__ dst,
                                int Kreal, int Ncols, int KT) {
    const int NT    = Ncols >> 4;
    const int total = KT * NT * 512;
    const int idx   = blockIdx.x * blockDim.x + threadIdx.x;
    if (idx >= total) return;
    const int e    = idx & 15;
    const int lane = (idx >> 4) & 31;
    const int tile = idx >> 9;            // kt*NT + nt
    const int nt   = tile % NT;
    const int kt   = tile / NT;
    const int k = kt * 32 + ((lane < 16) ? 0 : 8) + e + ((e < 8) ? 0 : 8);
    const int n = nt * 16 + (lane & 15);
    const float v = (k < Kreal) ? src[(size_t)k * Ncols + n] : 0.0f;
    dst[idx] = (half_t)v;
}

__global__ void gnnca_zero_f32(float* __restrict__ p, long long n) {
    long long i = (long long)blockIdx.x * blockDim.x + threadIdx.x;
    if (i < n) p[i] = 0.0f;
}

// ---------------------------------------------------------------------------
// Kernel 1: per-node MLP.  One wave = one 16-node tile.
// pre = relu(relu(relu(x@W1+b1)@W2+b2)@W3+b3);  msg = relu(pre@Wl+bl)
// pre, msg stored f16 (halves edge-gather bandwidth on the 23.3 TB/s HBM).
// ---------------------------------------------------------------------------
__global__ __launch_bounds__(WAVES * 32) void gnnca_pre_kernel(
    const float* __restrict__ x,
    const float* __restrict__ b1, const float* __restrict__ b2,
    const float* __restrict__ b3, const float* __restrict__ bl,
    const half_t* __restrict__ W1f, const half_t* __restrict__ W2f,
    const half_t* __restrict__ W3f, const half_t* __restrict__ Wlf,
    half_t* __restrict__ preH, half_t* __restrict__ msgH, int num_tiles)
{
    __shared__ half_t lds[WAVES][2][16 * H];
    const int wave = threadIdx.x >> 5;
    const int lane = threadIdx.x & 31;
    const int tile = blockIdx.x * WAVES + wave;
    if (tile >= num_tiles) return;
    half_t* A0 = &lds[wave][0][0];
    half_t* A1 = &lds[wave][1][0];
    const int m0 = tile * 16;

    // layer 1: K=16 (padded to 32) -> A fragment built straight from global f32 x
    {
        V16H a;
        const int m = lane & 15;
        const float* row = x + (size_t)(m0 + m) * DIM + ((lane < 16) ? 0 : 8);
#pragma unroll
        for (int e = 0; e < 8; ++e) { a.h[0][e] = (half_t)row[e]; a.h[1][e] = (half_t)0.0f; }
#pragma unroll
        for (int nt = 0; nt < 8; ++nt) {
            v8f c = {};
            c = wmma_f16(a.v, load_B_frag(W1f, 0, nt, 8, lane), c);
            store_C_lds(A0, lane, nt, H, c, b1, true);
        }
        asm volatile("s_wait_dscnt 0" ::: "memory");
    }

    gemm_layer<4, 8>(A0, A1, W2f, b2, lane, H, H, true);
    gemm_layer<4, 8>(A1, A0, W3f, b3, lane, H, H, true);      // pre -> A0

    {   // pre tile -> global f16 (contiguous 2048 halves)
        const v8h* sp = (const v8h*)A0;
        v8h* dp = (v8h*)(preH + (size_t)m0 * H);
#pragma unroll
        for (int j = lane; j < 256; j += 32) dp[j] = sp[j];
    }

    gemm_layer<4, 8>(A0, A1, Wlf, bl, lane, H, H, true);      // msg -> A1

    {   // msg tile -> global f16
        const v8h* sp = (const v8h*)A1;
        v8h* dp = (v8h*)(msgH + (size_t)m0 * H);
#pragma unroll
        for (int j = lane; j < 256; j += 32) dp[j] = sp[j];
    }
}

// ---------------------------------------------------------------------------
// Kernel 2: edge scatter-sum.  thread = (edge, 4-channel chunk):
// one 8-byte f16 gather + 4 f32 global atomics.
// ---------------------------------------------------------------------------
__global__ __launch_bounds__(256) void gnnca_scatter_kernel(
    const int* __restrict__ ei, const half_t* __restrict__ msgH,
    float* __restrict__ aggF)
{
    const long long tid = (long long)blockIdx.x * blockDim.x + threadIdx.x;
    if (tid >= (long long)N_EDGES * 32) return;
    const int e     = (int)(tid >> 5);
    const int chunk = (int)(tid & 31);
    const int s = ei[e];
    const int d = ei[N_EDGES + e];
    const v4h mv = *(const v4h*)(msgH + (size_t)s * H + chunk * 4);
    float* ap = aggF + (size_t)d * H + chunk * 4;
    atomicAdd(ap + 0, (float)mv[0]);
    atomicAdd(ap + 1, (float)mv[1]);
    atomicAdd(ap + 2, (float)mv[2]);
    atomicAdd(ap + 3, (float)mv[3]);
}

// ---------------------------------------------------------------------------
// Kernel 3: post MLP on concat([pre, agg]) -> out[N,16], outer ReLU applied.
// ---------------------------------------------------------------------------
__global__ __launch_bounds__(WAVES * 32) void gnnca_post_kernel(
    const half_t* __restrict__ preH, const float* __restrict__ aggF,
    const float* __restrict__ pb1, const float* __restrict__ pb2,
    const float* __restrict__ pb3,
    const half_t* __restrict__ P1f, const half_t* __restrict__ P2f,
    const half_t* __restrict__ P3f,
    float* __restrict__ out, int num_tiles)
{
    __shared__ half_t ldsA[WAVES][16 * 256];
    __shared__ half_t ldsB[WAVES][16 * H];
    const int wave = threadIdx.x >> 5;
    const int lane = threadIdx.x & 31;
    const int tile = blockIdx.x * WAVES + wave;
    if (tile >= num_tiles) return;
    half_t* A = &ldsA[wave][0];
    half_t* B = &ldsB[wave][0];
    const int m0 = tile * 16;

    // build concat tile [16 x 256] f16 in LDS: cols 0-127 = pre, 128-255 = agg
    {
        const v8h* ps = (const v8h*)(preH + (size_t)m0 * H);
        const float* as = aggF + (size_t)m0 * H;
#pragma unroll
        for (int j = lane; j < 256; j += 32) {
            const int m = j >> 4, c8 = j & 15;
            *(v8h*)(A + m * 256 + c8 * 8) = ps[j];
            const float* f = as + j * 8;
            v8h hv;
#pragma unroll
            for (int ee = 0; ee < 8; ++ee) hv[ee] = (half_t)f[ee];
            *(v8h*)(A + m * 256 + 128 + c8 * 8) = hv;
        }
        asm volatile("s_wait_dscnt 0" ::: "memory");
    }

    gemm_layer<8, 8>(A, B, P1f, pb1, lane, 256, H, true);
    gemm_layer<4, 8>(B, A, P2f, pb2, lane, H,   H, true);   // h2 -> A (16x128)

    // final: out = relu(h2 @ P3 + pb3), single 16-wide N tile
    {
        v8f c = {};
#pragma unroll
        for (int kt = 0; kt < 4; ++kt) {
            v16h a = load_A_lds(A, lane, kt * 32, H);
            c = wmma_f16(a, load_B_frag(P3f, kt, 0, 1, lane), c);
        }
        const int n     = lane & 15;
        const int mbase = (lane < 16) ? 0 : 8;
        const float bv  = pb3[n];
#pragma unroll
        for (int r = 0; r < 8; ++r)
            out[(size_t)(m0 + mbase + r) * DIM + n] = fmaxf(c[r] + bv, 0.0f);
    }
}

// ---------------------------------------------------------------------------
extern "C" void kernel_launch(void* const* d_in, const int* in_sizes, int n_in,
                              void* d_out, int out_size, void* d_ws, size_t ws_size,
                              hipStream_t stream) {
    (void)in_sizes; (void)n_in; (void)out_size; (void)ws_size;
    const float* x   = (const float*)d_in[0];
    const int*   ei  = (const int*)d_in[1];
    const float* W1  = (const float*)d_in[2];
    const float* b1  = (const float*)d_in[3];
    const float* W2  = (const float*)d_in[4];
    const float* b2  = (const float*)d_in[5];
    const float* W3  = (const float*)d_in[6];
    const float* b3  = (const float*)d_in[7];
    const float* Wl  = (const float*)d_in[8];
    const float* bl  = (const float*)d_in[9];
    const float* P1  = (const float*)d_in[10];
    const float* pb1 = (const float*)d_in[11];
    const float* P2  = (const float*)d_in[12];
    const float* pb2 = (const float*)d_in[13];
    const float* P3  = (const float*)d_in[14];
    const float* pb3 = (const float*)d_in[15];

    // workspace layout
    char* ws = (char*)d_ws;
    size_t off = 0;
    half_t* preH = (half_t*)(ws + off); off += (size_t)N_NODES * H * sizeof(half_t);
    half_t* msgH = (half_t*)(ws + off); off += (size_t)N_NODES * H * sizeof(half_t);
    float*  aggF = (float*) (ws + off); off += (size_t)N_NODES * H * sizeof(float);
    half_t* W1f = (half_t*)(ws + off); off += (size_t)1 * 8 * 512 * sizeof(half_t);
    half_t* W2f = (half_t*)(ws + off); off += (size_t)4 * 8 * 512 * sizeof(half_t);
    half_t* W3f = (half_t*)(ws + off); off += (size_t)4 * 8 * 512 * sizeof(half_t);
    half_t* Wlf = (half_t*)(ws + off); off += (size_t)4 * 8 * 512 * sizeof(half_t);
    half_t* P1f = (half_t*)(ws + off); off += (size_t)8 * 8 * 512 * sizeof(half_t);
    half_t* P2f = (half_t*)(ws + off); off += (size_t)4 * 8 * 512 * sizeof(half_t);
    half_t* P3f = (half_t*)(ws + off); off += (size_t)4 * 1 * 512 * sizeof(half_t);

    auto sw = [&](const float* s, half_t* dfr, int Kreal, int Ncols, int KT) {
        const int tot = KT * (Ncols / 16) * 512;
        gnnca_swizzle_w<<<(tot + 255) / 256, 256, 0, stream>>>(s, dfr, Kreal, Ncols, KT);
    };
    sw(W1, W1f,  16, 128, 1);
    sw(W2, W2f, 128, 128, 4);
    sw(W3, W3f, 128, 128, 4);
    sw(Wl, Wlf, 128, 128, 4);
    sw(P1, P1f, 256, 128, 8);
    sw(P2, P2f, 128, 128, 4);
    sw(P3, P3f, 128,  16, 4);

    const long long aggN = (long long)N_NODES * H;
    gnnca_zero_f32<<<(int)((aggN + 255) / 256), 256, 0, stream>>>(aggF, aggN);

    const int tiles   = (N_NODES + 15) / 16;                  // 6250
    const int mblocks = (tiles + WAVES - 1) / WAVES;

    gnnca_pre_kernel<<<mblocks, WAVES * 32, 0, stream>>>(
        x, b1, b2, b3, bl, W1f, W2f, W3f, Wlf, preH, msgH, tiles);

    const long long sthreads = (long long)N_EDGES * 32;
    gnnca_scatter_kernel<<<(int)((sthreads + 255) / 256), 256, 0, stream>>>(ei, msgH, aggF);

    gnnca_post_kernel<<<mblocks, WAVES * 32, 0, stream>>>(
        preH, aggF, pb1, pb2, pb3, P1f, P2f, P3f, (float*)d_out, tiles);
}
